// RNN_36318243455554
// MI455X (gfx1250) — matrix operand order repaired and checked by reading, not compiled, not run on hardware
//
#include <hip/hip_runtime.h>

// RNN: h_t = tanh(W_h h_{t-1} + b_h + W_in x_t + b_in); T=1024, B=64, F=H=1024.
// Phase 1: x_proj = x @ W_in^T + b_in (bf16 WMMA GEMM) written into d_out.
// Phase 2: 16 persistent WGs, each owns 64 hidden columns. W_h slice LDS-resident.
//   TDM variant: bf16 W_h + double-buffered bf16 h live in d_ws; Tensor Data Mover
//   (tensor_load_to_lds, TENSORcnt) DMAs tiles into LDS, overlapped with the
//   x_proj accumulator init. Fallback variant converts fp32->bf16 with VALU.

typedef __attribute__((ext_vector_type(16))) __bf16        v16bf;
typedef __attribute__((ext_vector_type(8)))  float         v8f;
typedef __attribute__((ext_vector_type(4)))  unsigned int  v4u;
typedef __attribute__((ext_vector_type(8)))  int           v8i;
typedef __attribute__((ext_vector_type(4)))  int           v4i;

#define SEQ_T   1024
#define BATCH_B 64
#define HID_H   1024
#define BH      (BATCH_B * HID_H)          // 65536
#define SEQ_ELEMS ((size_t)SEQ_T * BH)     // 67108864

// TDM LDS row stride: 2048 B of data + 2x16 B pad (pad_interval=256 dw, amount=4 dw)
#define TROW 2080
// fallback LDS row stride (conflict-free for b128 frag reads)
#define FROW 2064

// fp32 -> packed 2x bf16, round-to-nearest-even
__device__ __forceinline__ unsigned int pack_bf16(float a, float b) {
    unsigned int ua = __float_as_uint(a), ub = __float_as_uint(b);
    ua = (ua + 0x7FFFu + ((ua >> 16) & 1u)) >> 16;
    ub = (ub + 0x7FFFu + ((ub >> 16) & 1u)) >> 16;
    return (ua & 0xFFFFu) | (ub << 16);
}
__device__ __forceinline__ unsigned short cvt_bf16(float f) {
    unsigned int u = __float_as_uint(f);
    u = (u + 0x7FFFu + ((u >> 16) & 1u)) >> 16;
    return (unsigned short)u;
}

__device__ __forceinline__ v16bf make_frag(const unsigned char* p0, const unsigned char* p1) {
    union { uint4 q[2]; v16bf v; } u;
    u.q[0] = *(const uint4*)p0;
    u.q[1] = *(const uint4*)p1;
    return u.v;
}

// Tensor Data Mover: 2D bf16 tile (rows x cols, row stride = cols) -> LDS with
// 16 B pad per 1024 B (LDS row stride = 2*cols + 32 when cols=1024).
__device__ __forceinline__ void tdm_load_2d_bf16(unsigned lds_addr, const void* gsrc,
                                                 unsigned rows, unsigned cols) {
    unsigned long long ga = (unsigned long long)(size_t)gsrc;
    v4u g0 = { 1u,                                    // count=1 (valid user D#)
               lds_addr,                              // LDS byte address
               (unsigned)(ga & 0xFFFFFFFFu),          // global_addr[31:0]
               ((unsigned)(ga >> 32) & 0x01FFFFFFu) | (2u << 30) };  // [56:32] | type=2
    v8i g1 = { (int)((1u << 16) | (1u << 20) | (7u << 22) | (3u << 25)),
               //      data_size=2B  pad_en     interval=256dw  amount=4dw
               (int)(cols << 16),                     // tensor_dim0[15:0] @ bits[63:48]
               (int)(rows << 16),                     // tensor_dim1[15:0] @ bits[95:80]
               (int)(cols << 16),                     // tile_dim0 @ bits[127:112]
               (int)rows,                             // tile_dim1 @ [143:128], tile_dim2=0
               (int)cols,                             // tensor_dim0_stride[31:0]
               0, 0 };
    v4i g2 = { 0, 0, 0, 0 };
    v4i g3 = { 0, 0, 0, 0 };
    v8i g4 = { 0, 0, 0, 0, 0, 0, 0, 0 };              // extra SGPR group (6-arg builtin)
    __builtin_amdgcn_tensor_load_to_lds(g0, g1, g2, g3, g4, 0);
}

// ---------------- Phase 0: W_h fp32 -> bf16 in workspace ----------------
__global__ __launch_bounds__(256)
void cvt_wh_kernel(const float* __restrict__ Wh, unsigned int* __restrict__ whbf) {
    const int g = blockIdx.x * 256 + threadIdx.x;   // 512 blocks -> 131072 threads
    #pragma unroll
    for (int i = 0; i < 4; ++i) {
        const int p = g + i * 131072;               // 524288 bf16 pairs total
        const float2 w = *(const float2*)(Wh + (size_t)p * 2);
        whbf[p] = pack_bf16(w.x, w.y);
    }
}

// ---------------- Phase 1: x_proj GEMM ----------------
// Block tile 128(M) x 64(N), K-step 32; 8 waves in 4(M) x 2(N), 2x2 subtiles each.
__global__ __launch_bounds__(256)
void xproj_kernel(const float* __restrict__ x, const float* __restrict__ Win,
                  const float* __restrict__ bin, float* __restrict__ out,
                  unsigned int* __restrict__ ctr)
{
    if (blockIdx.x == 0 && blockIdx.y == 0 && threadIdx.x == 0) *ctr = 0u;

    __shared__ __align__(16) unsigned char sA[128 * 80];
    __shared__ __align__(16) unsigned char sB[64 * 80];

    const int tid  = threadIdx.x;
    const int lane = tid & 31, wave = tid >> 5;
    const int half = lane >> 4, l16 = lane & 15;
    const int wm = wave & 3, wn = wave >> 2;
    const size_t m0 = (size_t)blockIdx.x * 128;
    const int    n0 = blockIdx.y * 64;

    v8f zero = {};
    v8f acc[2][2] = {{zero, zero}, {zero, zero}};

    for (int kt = 0; kt < 32; ++kt) {
        const int k0 = kt * 32;
        #pragma unroll
        for (int i = 0; i < 4; ++i) {
            int f = tid + i * 256; int row = f >> 3; int c4 = (f & 7) * 4;
            const float4 xv = *(const float4*)(x + (m0 + row) * 1024 + k0 + c4);
            uint2 p; p.x = pack_bf16(xv.x, xv.y); p.y = pack_bf16(xv.z, xv.w);
            *(uint2*)(sA + row * 80 + c4 * 2) = p;
        }
        #pragma unroll
        for (int i = 0; i < 2; ++i) {
            int f = tid + i * 256; int row = f >> 3; int c4 = (f & 7) * 4;
            const float4 wv = *(const float4*)(Win + (size_t)(n0 + row) * 1024 + k0 + c4);
            uint2 p; p.x = pack_bf16(wv.x, wv.y); p.y = pack_bf16(wv.z, wv.w);
            *(uint2*)(sB + row * 80 + c4 * 2) = p;
        }
        __syncthreads();

        v16bf a[2], b[2];
        #pragma unroll
        for (int mi = 0; mi < 2; ++mi) {
            const unsigned char* base = sA + (32 * wm + 16 * mi + l16) * 80 + half * 16;
            a[mi] = make_frag(base, base + 32);
        }
        #pragma unroll
        for (int ni = 0; ni < 2; ++ni) {
            const unsigned char* base = sB + (32 * wn + 16 * ni + l16) * 80 + half * 16;
            b[ni] = make_frag(base, base + 32);
        }
        #pragma unroll
        for (int mi = 0; mi < 2; ++mi)
            #pragma unroll
            for (int ni = 0; ni < 2; ++ni)
                acc[mi][ni] = __builtin_amdgcn_wmma_f32_16x16x32_bf16(
                    false, a[mi], false, b[ni], (short)0, acc[mi][ni], false, false);
        __syncthreads();
    }

    #pragma unroll
    for (int ni = 0; ni < 2; ++ni) {
        const int gcol = n0 + 32 * wn + 16 * ni + l16;
        const float bias = bin[gcol];
        #pragma unroll
        for (int mi = 0; mi < 2; ++mi) {
            const size_t gm = m0 + 32 * wm + 16 * mi + 8 * half;
            #pragma unroll
            for (int r = 0; r < 8; ++r)
                out[(gm + r) * 1024 + gcol] = acc[mi][ni][r] + bias;
        }
    }
}

// ---------------- Phase 2 (TDM variant) ----------------
__global__ __launch_bounds__(256)
void rnn_step_tdm(const unsigned short* __restrict__ whbf, const float* __restrict__ bh,
                  float* __restrict__ out, unsigned int* __restrict__ ctr,
                  unsigned short* __restrict__ hbf0, unsigned short* __restrict__ hbf1)
{
    extern __shared__ __align__(16) unsigned char smem[];
    unsigned char* Wsh = smem;                 // 64 x TROW
    unsigned char* Hs  = smem + 64 * TROW;     // 64 x TROW

    const int tid  = threadIdx.x;
    const int lane = tid & 31, wave = tid >> 5;
    const int half = lane >> 4, l16 = lane & 15;
    const int mt  = wave & 3;
    const int ntb = (wave >> 2) * 2;
    const int i0  = blockIdx.x * 64;

    // prologue: TDM-load W_h bf16 slice into LDS (resident for all 1024 steps)
    if (wave == 0 && lane == 0)
        tdm_load_2d_bf16((unsigned)(size_t)Wsh, whbf + (size_t)i0 * 1024, 64, 1024);
    float bias[2];
    #pragma unroll
    for (int ni = 0; ni < 2; ++ni) bias[ni] = bh[i0 + (ntb + ni) * 16 + l16];
    if (wave == 0) __builtin_amdgcn_s_wait_tensorcnt(0);
    __syncthreads();

    for (int t = 0; t < SEQ_T; ++t) {
        float* outT = out + (size_t)t * BH;

        // kick off the async h_{t-1} tile DMA first, then overlap with xp init
        if (t > 0 && wave == 0 && lane == 0) {
            const unsigned short* hsrc = ((t & 1) == 0) ? hbf1 : hbf0;  // (t-1)&1
            tdm_load_2d_bf16((unsigned)(size_t)Hs, hsrc, 64, 1024);
        }

        v8f acc[2];
        #pragma unroll
        for (int ni = 0; ni < 2; ++ni) {
            const int col = i0 + (ntb + ni) * 16 + l16;
            #pragma unroll
            for (int r = 0; r < 8; ++r) {
                const int m = 16 * mt + 8 * half + r;
                acc[ni][r] = outT[(size_t)m * 1024 + col] + bias[ni];
            }
        }

        if (t > 0) {
            if (wave == 0) __builtin_amdgcn_s_wait_tensorcnt(0);
            __syncthreads();
            #pragma unroll 2
            for (int kt = 0; kt < 32; ++kt) {
                const int cpad = (kt >= 16) ? 16 : 0;   // TDM mid-row pad after 1024 B
                const unsigned char* baseA =
                    Hs + (16 * mt + l16) * TROW + kt * 64 + cpad + half * 16;
                const v16bf a = make_frag(baseA, baseA + 32);
                #pragma unroll
                for (int ni = 0; ni < 2; ++ni) {
                    const unsigned char* baseB =
                        Wsh + ((ntb + ni) * 16 + l16) * TROW + kt * 64 + cpad + half * 16;
                    const v16bf b = make_frag(baseB, baseB + 32);
                    acc[ni] = __builtin_amdgcn_wmma_f32_16x16x32_bf16(
                        false, a, false, b, (short)0, acc[ni], false, false);
                }
            }
        }

        // epilogue: tanh; fp32 -> d_out, bf16 -> double-buffered scratch for TDM
        unsigned short* hdst = ((t & 1) == 0) ? hbf0 : hbf1;
        #pragma unroll
        for (int ni = 0; ni < 2; ++ni) {
            const int col = i0 + (ntb + ni) * 16 + l16;
            #pragma unroll
            for (int r = 0; r < 8; ++r) {
                const int m = 16 * mt + 8 * half + r;
                const float hv = tanhf(acc[ni][r]);
                outT[(size_t)m * 1024 + col] = hv;
                hdst[(size_t)m * 1024 + col] = cvt_bf16(hv);
                if (t == SEQ_T - 1)
                    out[SEQ_ELEMS + (size_t)m * 1024 + col] = hv;
            }
        }

        if (t < SEQ_T - 1) {
            __threadfence();
            __syncthreads();
            if (tid == 0) {
                __hip_atomic_fetch_add(ctr, 1u, __ATOMIC_RELEASE, __HIP_MEMORY_SCOPE_AGENT);
                const unsigned int target = 16u * (unsigned int)(t + 1);
                while (__hip_atomic_load(ctr, __ATOMIC_ACQUIRE, __HIP_MEMORY_SCOPE_AGENT) < target)
                    __builtin_amdgcn_s_sleep(8);
            }
            __syncthreads();
        }
    }
}

// ---------------- Phase 2 (fallback, VALU conversion staging) ----------------
__global__ __launch_bounds__(256)
void rnn_step_fallback(const float* __restrict__ Wh, const float* __restrict__ bh,
                       float* __restrict__ out, unsigned int* __restrict__ ctr)
{
    extern __shared__ __align__(16) unsigned char smem[];
    unsigned char* Wsh = smem;
    unsigned char* Hs  = smem + 64 * FROW;

    const int tid  = threadIdx.x;
    const int lane = tid & 31, wave = tid >> 5;
    const int half = lane >> 4, l16 = lane & 15;
    const int mt  = wave & 3;
    const int ntb = (wave >> 2) * 2;
    const int i0  = blockIdx.x * 64;

    #pragma unroll 4
    for (int it = 0; it < 128; ++it) {
        int d = tid + it * 256; int row = d >> 9; int cp = d & 511;
        const float2 w = *(const float2*)(Wh + (size_t)(i0 + row) * 1024 + cp * 2);
        *(unsigned int*)(Wsh + row * FROW + cp * 4) = pack_bf16(w.x, w.y);
    }
    float bias[2];
    #pragma unroll
    for (int ni = 0; ni < 2; ++ni) bias[ni] = bh[i0 + (ntb + ni) * 16 + l16];
    __syncthreads();

    for (int t = 0; t < SEQ_T; ++t) {
        float* outT = out + (size_t)t * BH;
        v8f acc[2];
        #pragma unroll
        for (int ni = 0; ni < 2; ++ni) {
            const int col = i0 + (ntb + ni) * 16 + l16;
            #pragma unroll
            for (int r = 0; r < 8; ++r) {
                const int m = 16 * mt + 8 * half + r;
                acc[ni][r] = outT[(size_t)m * 1024 + col] + bias[ni];
            }
        }
        if (t > 0) {
            const float* hprev = outT - BH;
            #pragma unroll 4
            for (int it = 0; it < 128; ++it) {
                int d = tid + it * 256; int row = d >> 9; int cp = d & 511;
                const float2 hv = *(const float2*)(hprev + (size_t)row * 1024 + cp * 2);
                *(unsigned int*)(Hs + row * FROW + cp * 4) = pack_bf16(hv.x, hv.y);
            }
            __syncthreads();
            #pragma unroll 2
            for (int kt = 0; kt < 32; ++kt) {
                const unsigned char* baseA = Hs + (16 * mt + l16) * FROW + kt * 64 + half * 16;
                const v16bf a = make_frag(baseA, baseA + 32);
                #pragma unroll
                for (int ni = 0; ni < 2; ++ni) {
                    const unsigned char* baseB =
                        Wsh + ((ntb + ni) * 16 + l16) * FROW + kt * 64 + half * 16;
                    const v16bf b = make_frag(baseB, baseB + 32);
                    acc[ni] = __builtin_amdgcn_wmma_f32_16x16x32_bf16(
                        false, a, false, b, (short)0, acc[ni], false, false);
                }
            }
        }
        #pragma unroll
        for (int ni = 0; ni < 2; ++ni) {
            const int col = i0 + (ntb + ni) * 16 + l16;
            #pragma unroll
            for (int r = 0; r < 8; ++r) {
                const int m = 16 * mt + 8 * half + r;
                const float hv = tanhf(acc[ni][r]);
                outT[(size_t)m * 1024 + col] = hv;
                if (t == SEQ_T - 1)
                    out[SEQ_ELEMS + (size_t)m * 1024 + col] = hv;
            }
        }
        if (t < SEQ_T - 1) {
            __threadfence();
            __syncthreads();
            if (tid == 0) {
                __hip_atomic_fetch_add(ctr, 1u, __ATOMIC_RELEASE, __HIP_MEMORY_SCOPE_AGENT);
                const unsigned int target = 16u * (unsigned int)(t + 1);
                while (__hip_atomic_load(ctr, __ATOMIC_ACQUIRE, __HIP_MEMORY_SCOPE_AGENT) < target)
                    __builtin_amdgcn_s_sleep(8);
            }
            __syncthreads();
        }
    }
}

extern "C" void kernel_launch(void* const* d_in, const int* in_sizes, int n_in,
                              void* d_out, int out_size, void* d_ws, size_t ws_size,
                              hipStream_t stream)
{
    (void)in_sizes; (void)n_in; (void)out_size;
    const float* x   = (const float*)d_in[0];
    const float* Win = (const float*)d_in[1];
    const float* bin = (const float*)d_in[2];
    const float* Wh  = (const float*)d_in[3];
    const float* bh  = (const float*)d_in[4];
    float* out = (float*)d_out;
    char* ws = (char*)d_ws;
    unsigned int* ctr = (unsigned int*)ws;

    // workspace layout: [ctr | pad to 256 | W_h bf16 (2 MB) | h bf16 x2 (128 KB each)]
    const size_t WHBF_OFF = 256;
    const size_t WHBF_BYTES = (size_t)HID_H * HID_H * 2;
    const size_t HBF_BYTES  = (size_t)BATCH_B * HID_H * 2;
    const size_t HBF0_OFF = WHBF_OFF + WHBF_BYTES;
    const size_t HBF1_OFF = HBF0_OFF + HBF_BYTES;
    const size_t WS_NEED  = HBF1_OFF + HBF_BYTES;

    xproj_kernel<<<dim3(512, 16), dim3(256), 0, stream>>>(x, Win, bin, out, ctr);

    if (ws_size >= WS_NEED) {
        cvt_wh_kernel<<<dim3(512), dim3(256), 0, stream>>>(Wh, (unsigned int*)(ws + WHBF_OFF));
        rnn_step_tdm<<<dim3(16), dim3(256), 2 * 64 * TROW, stream>>>(
            (const unsigned short*)(ws + WHBF_OFF), bh, out, ctr,
            (unsigned short*)(ws + HBF0_OFF), (unsigned short*)(ws + HBF1_OFF));
    } else {
        rnn_step_fallback<<<dim3(16), dim3(256), 2 * 64 * FROW, stream>>>(Wh, bh, out, ctr);
    }
}
